// UltimateRAv4_23974507446663
// MI455X (gfx1250) — matrix operand order repaired and checked by reading, not compiled, not run on hardware
//
#include <hip/hip_runtime.h>
#include <hip/hip_bf16.h>

// ---------------------------------------------------------------------------
// CDNA5 (gfx1250) bf16 WMMA pipeline with TDM (tensor_load_to_lds) staging:
//   cvt(x) + transpose-cvt(Wa*colscale, Wp) -> GEMM1 (wmma bf16, TDM double
//   buffered) -> routing/swap -> flash attn (wmma bf16, TDM K-chunk, online
//   softmax) -> GEMM2 (wmma bf16, f32 out)
// ---------------------------------------------------------------------------

typedef __attribute__((ext_vector_type(16))) __bf16 v16bf;
typedef __attribute__((ext_vector_type(8)))  float  v8f;

union BF16Frag { uint4 u[2]; v16bf v; };

static __device__ __forceinline__ unsigned short bf16_bits(float x) {
    __bf16 b = (__bf16)x;
    return __builtin_bit_cast(unsigned short, b);
}

template <int PAT>
static __device__ __forceinline__ float swz(float v) {
    return __int_as_float(__builtin_amdgcn_ds_swizzle(__float_as_int(v), PAT));
}
// group-of-32 xor-swizzle patterns: (xor<<10) | and_mask(0x1F)
#define SWZ_X1  0x041F
#define SWZ_X2  0x081F
#define SWZ_X4  0x101F
#define SWZ_X8  0x201F

// ---------------------------------------------------------------------------
// TDM support (probe-verified builtin; fall back to manual copies if absent)
// ---------------------------------------------------------------------------
#if defined(__has_builtin)
#if __has_builtin(__builtin_amdgcn_tensor_load_to_lds)
#define HAVE_TDM 1
#endif
#if __has_builtin(__builtin_amdgcn_s_wait_tensorcnt)
#define HAVE_TCNT 1
#endif
#endif

static __device__ __forceinline__ void wait_tensor0() {
#if defined(HAVE_TCNT)
    __builtin_amdgcn_s_wait_tensorcnt(0);
#else
    asm volatile("s_wait_tensorcnt 0x0" ::: "memory");
#endif
}

static __device__ __forceinline__ unsigned int lds_addr_of(const void* p) {
    // flat address of an LDS object: low 32 bits are the LDS byte offset
    return (unsigned int)(unsigned long long)(size_t)p;
}

#if defined(HAVE_TDM)
typedef __attribute__((ext_vector_type(4))) unsigned int v4u;
typedef __attribute__((ext_vector_type(8))) int v8i;
typedef __attribute__((ext_vector_type(4))) int v4i;

// 2D tile load global->LDS with per-row LDS padding.
//  dsz_code: 0=1B,1=2B,2=4B,3=8B (element units for dims/strides)
//  pad codes per D# spec: interval 3 -> 16 dwords, 5 -> 64 dwords; amount 3 -> 4 dwords
static __device__ __forceinline__ void tdm_load_2d(const void* gsrc, unsigned int lds_addr,
                                                   int dsz_code, int width_elems, int rows,
                                                   int row_stride_elems,
                                                   int pad_interval_code, int pad_amount_code) {
    unsigned long long ga = (unsigned long long)(size_t)gsrc;
    v4u g0 = { 1u,                                    // count=1, user mode
               lds_addr,                              // LDS byte address
               (unsigned int)ga,                      // global_addr[31:0]
               (unsigned int)((ga >> 32) & 0x01FFFFFFull) | 0x80000000u }; // addr[56:32] | type=2
    int d0 = (dsz_code << 16) | (1 << 20) | (pad_interval_code << 22) | (pad_amount_code << 25);
    v8i g1 = { d0,
               (int)((unsigned)width_elems << 16),                              // tensor_dim0 lo16
               (int)(((unsigned)width_elems >> 16) | ((unsigned)rows << 16)),   // dim0 hi | dim1 lo
               (int)(((unsigned)rows >> 16) | ((unsigned)width_elems << 16)),   // dim1 hi | tile_dim0
               rows,                                                            // tile_dim1 (tile_dim2=0)
               row_stride_elems,                                                // dim0_stride lo32
               0, 0 };                                                          // stride hi, dim1_stride
    v4i z = { 0, 0, 0, 0 };
#if __clang_major__ >= 23
    v8i z8 = { 0, 0, 0, 0, 0, 0, 0, 0 };
    __builtin_amdgcn_tensor_load_to_lds(g0, g1, z, z, z8, 0);
#else
    __builtin_amdgcn_tensor_load_to_lds(g0, g1, z, z, 0);
#endif
}
#endif  // HAVE_TDM

// ---------------------------------------------------------------------------
// Elementwise converters
// ---------------------------------------------------------------------------
__global__ __launch_bounds__(256) void cvt_f32_to_bf16_v4(const float* __restrict__ in,
                                                          __bf16* __restrict__ out, int n4) {
    int i = blockIdx.x * 256 + threadIdx.x;
    if (i >= n4) return;
    float4 v = reinterpret_cast<const float4*>(in)[i];
    uint2 r;
    r.x = (unsigned int)bf16_bits(v.x) | ((unsigned int)bf16_bits(v.y) << 16);
    r.y = (unsigned int)bf16_bits(v.z) | ((unsigned int)bf16_bits(v.w) << 16);
    reinterpret_cast<uint2*>(out)[i] = r;
}

// Transpose-convert f32[R][Ccols] -> bf16[Ccols][R]; optional Wb column scaling.
template <bool SCALED>
__global__ __launch_bounds__(256) void cvt_transpose_bf16(const float* __restrict__ in,
                                                          __bf16* __restrict__ out,
                                                          int R, int Ccols,
                                                          const float* __restrict__ w_std,
                                                          const float* __restrict__ w_rec) {
    __shared__ float tile[32][33];
    const int c0 = blockIdx.x * 32;
    const int r0 = blockIdx.y * 32;
    const int tr = threadIdx.x >> 5;   // 0..7
    const int tc = threadIdx.x & 31;   // 0..31

    float sc = 1.0f;
    if (SCALED) {
        int c = c0 + tc;
        if (c < 3840)      sc = sqrtf(fmaxf(w_std[(c % 1920) / 120], 1e-8f));
        else if (c < 5888) sc = 1.0f;
        else               sc = sqrtf(fmaxf(w_rec[((c - 5888) & 127) >> 3], 1e-8f));
    }
#pragma unroll
    for (int j = 0; j < 4; ++j) {
        int r = tr + j * 8;
        tile[r][tc] = in[(size_t)(r0 + r) * Ccols + c0 + tc] * sc;
    }
    __syncthreads();
#pragma unroll
    for (int j = 0; j < 4; ++j) {
        int r = tr + j * 8;            // output row within tile = input column
        out[(size_t)(c0 + r) * R + r0 + tc] = (__bf16)tile[tc][r];
    }
}

// ---------------------------------------------------------------------------
// bf16 WMMA GEMM: C[M,N] = A[M,K] * BT[N,K]^T; A,BT row-major bf16 (K-major).
// Block tile 128x128, 8 waves (2x4), wave tile 64x32, K step 32.
// TDM path: double-buffered LDS tiles, wave 0 drives the tensor DMA.
// ---------------------------------------------------------------------------
template <bool F32OUT>
__global__ __launch_bounds__(256) void gemm_wmma_bf16(const __bf16* __restrict__ A,
                                                      const __bf16* __restrict__ BT,
                                                      void* __restrict__ C,
                                                      int M, int N, int K) {
    __shared__ unsigned int As[2][128 * 20];   // [m][16 dwords of K-pairs], pitch 20
    __shared__ unsigned int Bs[2][128 * 20];   // [n][16 dwords of K-pairs], pitch 20

    const int tid   = threadIdx.x;
    const int lane  = tid & 31;
    const int wave  = tid >> 5;
    const int laneN = lane & 15;
    const int hl    = lane >> 4;
    const int waveM = wave >> 2;               // 0..1
    const int waveN = wave & 3;                // 0..3
    const int gm0   = blockIdx.y * 128;
    const int gn0   = blockIdx.x * 128;

    v8f zero = {0.f, 0.f, 0.f, 0.f, 0.f, 0.f, 0.f, 0.f};
    v8f acc[4][2];
#pragma unroll
    for (int mi = 0; mi < 4; ++mi)
#pragma unroll
        for (int ni = 0; ni < 2; ++ni) acc[mi][ni] = zero;

    const __bf16* Abase = A + (size_t)gm0 * K;
    const __bf16* Bbase = BT + (size_t)gn0 * K;
    const int KT = K >> 5;

#if defined(HAVE_TDM)
    if (wave == 0) {
        tdm_load_2d(Abase, lds_addr_of(&As[0][0]), 1, 32, 128, K, 3, 3);
        tdm_load_2d(Bbase, lds_addr_of(&Bs[0][0]), 1, 32, 128, K, 3, 3);
    }
#endif

    for (int kt = 0; kt < KT; ++kt) {
        const int cur = kt & 1;
#if defined(HAVE_TDM)
        if (wave == 0) wait_tensor0();         // tiles for kt have landed
        __syncthreads();
        if (kt + 1 < KT && wave == 0) {        // prefetch kt+1 into the other buffer
            tdm_load_2d(Abase + (kt + 1) * 32, lds_addr_of(&As[cur ^ 1][0]), 1, 32, 128, K, 3, 3);
            tdm_load_2d(Bbase + (kt + 1) * 32, lds_addr_of(&Bs[cur ^ 1][0]), 1, 32, 128, K, 3, 3);
        }
#else
        {
            const int row = tid >> 1, hf2 = tid & 1;
            const __bf16* ap = Abase + (size_t)row * K + kt * 32 + hf2 * 16;
            uint4 a0 = reinterpret_cast<const uint4*>(ap)[0];
            uint4 a1 = reinterpret_cast<const uint4*>(ap)[1];
            unsigned int* asw = &As[cur][row * 20 + hf2 * 8];
            *reinterpret_cast<uint4*>(asw)     = a0;
            *reinterpret_cast<uint4*>(asw + 4) = a1;
            const __bf16* bp = Bbase + (size_t)row * K + kt * 32 + hf2 * 16;
            uint4 b0 = reinterpret_cast<const uint4*>(bp)[0];
            uint4 b1 = reinterpret_cast<const uint4*>(bp)[1];
            unsigned int* bsw = &Bs[cur][row * 20 + hf2 * 8];
            *reinterpret_cast<uint4*>(bsw)     = b0;
            *reinterpret_cast<uint4*>(bsw + 4) = b1;
            if (kt + 1 < KT) {
                __builtin_prefetch((const void*)(ap + 32), 0, 0);
                __builtin_prefetch((const void*)(bp + 32), 0, 0);
            }
        }
        __syncthreads();
#endif
        // ---- fragments + WMMA ----
        BF16Frag afr[4];
#pragma unroll
        for (int mi = 0; mi < 4; ++mi) {
            const unsigned int* s = &As[cur][(waveM * 64 + mi * 16 + laneN) * 20 + hl * 4];
            afr[mi].u[0] = *reinterpret_cast<const uint4*>(s);
            afr[mi].u[1] = *reinterpret_cast<const uint4*>(s + 8);
        }
#pragma unroll
        for (int ni = 0; ni < 2; ++ni) {
            BF16Frag bfr;
            const unsigned int* s = &Bs[cur][(waveN * 32 + ni * 16 + laneN) * 20 + hl * 4];
            bfr.u[0] = *reinterpret_cast<const uint4*>(s);
            bfr.u[1] = *reinterpret_cast<const uint4*>(s + 8);
#pragma unroll
            for (int mi = 0; mi < 4; ++mi)
                acc[mi][ni] = __builtin_amdgcn_wmma_f32_16x16x32_bf16(
                    false, afr[mi].v, false, bfr.v, (short)0, acc[mi][ni], false, false);
        }
        __syncthreads();
    }

    // ---- epilogue: C layout lane holds M = i + 8*hl, N = laneN ----
#pragma unroll
    for (int mi = 0; mi < 4; ++mi)
#pragma unroll
        for (int ni = 0; ni < 2; ++ni)
#pragma unroll
            for (int i = 0; i < 8; ++i) {
                int row = gm0 + waveM * 64 + mi * 16 + i + 8 * hl;
                int col = gn0 + waveN * 32 + ni * 16 + laneN;
                float v = acc[mi][ni][i];
                if (F32OUT)
                    reinterpret_cast<float*>(C)[(size_t)row * N + col] = v;
                else
                    reinterpret_cast<__bf16*>(C)[(size_t)row * N + col] = (__bf16)v;
            }
}

// ---------------------------------------------------------------------------
// Routing: split fused [4096,6144] -> Qf/Kf/V [B,H,T,128] bf16, swap low-rank
// tails for heads with w_rec > 0.1; bake softmax scale into Qf.
// ---------------------------------------------------------------------------
__global__ __launch_bounds__(256) void routing_kernel(const __bf16* __restrict__ fused,
                                                      const float* __restrict__ w_rec,
                                                      __bf16* __restrict__ Qf,
                                                      __bf16* __restrict__ Kf,
                                                      __bf16* __restrict__ Vv) {
    const int tok = blockIdx.x;                // 0..4095
    const int b = tok >> 10, t = tok & 1023;
    const __bf16* f = fused + (size_t)tok * 6144;
    const float SCALE = 0.08838834764831845f;  // 1/sqrt(128)
    for (int c = threadIdx.x; c < 2048; c += 256) {
        int h = c >> 7, d = c & 127;
        float qv, kv;
        if (d < 120) {
            qv = (float)f[h * 120 + d];
            kv = (float)f[1920 + h * 120 + d];
        } else {
            int j = d - 120;
            float ql = (float)f[5888 + h * 8 + j];
            float kl = (float)f[6016 + h * 8 + j];
            bool ra = w_rec[h] > 0.1f;
            qv = ra ? kl : ql;
            kv = ra ? ql : kl;
        }
        size_t o = ((size_t)(b * 16 + h) * 1024 + t) * 128 + d;
        Qf[o] = (__bf16)(qv * SCALE);
        Kf[o] = (__bf16)kv;
        Vv[o] = (__bf16)((float)f[3840 + c]);
    }
}

// ---------------------------------------------------------------------------
// Flash attention: block = 64 queries, 4 waves, wave = 16 query rows.
// K chunk staged via TDM; V transposed manually (overlaps the DMA).
// ---------------------------------------------------------------------------
__global__ __launch_bounds__(128) void flash_attn_kernel(const __bf16* __restrict__ Qf,
                                                         const __bf16* __restrict__ Kf,
                                                         const __bf16* __restrict__ Vv,
                                                         __bf16* __restrict__ Out) {
    __shared__ unsigned int   Ks[64 * 68];     // K chunk [key][128 bf16], pitch 68 dw (TDM pad)
    __shared__ unsigned int   Vs[128 * 36];    // V chunk transposed [d][64 bf16], pitch 36 dw
    __shared__ unsigned short Ps[4 * 16 * 72]; // per-wave P staging [16][64], pitch 72 bf16

    const int tid   = threadIdx.x;
    const int lane  = tid & 31;
    const int wave  = tid >> 5;                // 0..3
    const int laneN = lane & 15;
    const int hl    = lane >> 4;
    const int qblk  = blockIdx.x;              // 0..15
    const int h     = blockIdx.y;
    const int b     = blockIdx.z;
    const int q0    = qblk * 64;
    const size_t bh = (size_t)(b * 16 + h) * 1024;

    // Q fragments: lane = query row (A layout), K-dim = d
    BF16Frag q[4];
    {
        const __bf16* qp = Qf + (bh + q0 + wave * 16 + laneN) * 128;
#pragma unroll
        for (int fr = 0; fr < 4; ++fr) {
            const uint4* p = reinterpret_cast<const uint4*>(qp + fr * 32 + hl * 8);
            q[fr].u[0] = p[0];
            q[fr].u[1] = p[2];  // +16 bf16
        }
    }

    v8f zero = {0.f, 0.f, 0.f, 0.f, 0.f, 0.f, 0.f, 0.f};
    v8f o[8];
    float mrow[8], lrow[8];
#pragma unroll
    for (int i = 0; i < 8; ++i) { o[i] = zero; mrow[i] = -3.0e38f; lrow[i] = 0.f; }

    const int nchunk = qblk + 1;
    for (int c = 0; c < nchunk; ++c) {
        const int k0 = c * 64;
        // ---- K chunk: TDM (8B elems, 32/row, 64 rows, pad 4dw per 64dw) ----
#if defined(HAVE_TDM)
        if (wave == 0)
            tdm_load_2d(Kf + (bh + k0) * 128, lds_addr_of(&Ks[0]), 3, 32, 64, 32, 5, 3);
#else
        for (int idx = tid; idx < 512; idx += 128) {
            int row = idx >> 3, seg = idx & 7;
            uint4 d4 = *reinterpret_cast<const uint4*>(Kf + (bh + k0 + row) * 128 + seg * 8);
            *reinterpret_cast<uint4*>(&Ks[row * 68 + seg * 4]) = d4;
        }
#endif
        // ---- V chunk transposed into [d][k] (overlaps TDM) ----
        for (int idx = tid; idx < 512; idx += 128) {
            int kp = idx >> 4, db = (idx & 15) * 8;
            const __bf16* vp = Vv + (bh + k0 + 2 * kp) * 128 + db;
            uint4 v0 = *reinterpret_cast<const uint4*>(vp);
            uint4 v1 = *reinterpret_cast<const uint4*>(vp + 128);
            const unsigned short* lo = reinterpret_cast<const unsigned short*>(&v0);
            const unsigned short* hi = reinterpret_cast<const unsigned short*>(&v1);
#pragma unroll
            for (int j = 0; j < 8; ++j)
                Vs[(db + j) * 36 + kp] = (unsigned int)lo[j] | ((unsigned int)hi[j] << 16);
        }
#if defined(HAVE_TDM)
        if (wave == 0) wait_tensor0();
#endif
        __syncthreads();

        // ---- S = Q * K^T : 4 n-tiles x 4 k-steps of wmma ----
        v8f s[4];
#pragma unroll
        for (int ni = 0; ni < 4; ++ni) {
            s[ni] = zero;
#pragma unroll
            for (int fr = 0; fr < 4; ++fr) {
                BF16Frag kb;
                const unsigned int* p = &Ks[(ni * 16 + laneN) * 68 + fr * 16 + hl * 4];
                kb.u[0] = *reinterpret_cast<const uint4*>(p);
                kb.u[1] = *reinterpret_cast<const uint4*>(p + 8);
                s[ni] = __builtin_amdgcn_wmma_f32_16x16x32_bf16(
                    false, q[fr].v, false, kb.v, (short)0, s[ni], false, false);
            }
        }

        // ---- causal mask (only the diagonal chunk) ----
        if (c == qblk) {
#pragma unroll
            for (int ni = 0; ni < 4; ++ni) {
                int kcol = k0 + ni * 16 + laneN;
#pragma unroll
                for (int i = 0; i < 8; ++i) {
                    int qrow = q0 + wave * 16 + i + 8 * hl;
                    if (kcol > qrow) s[ni][i] = -3.0e38f;
                }
            }
        }

        // ---- online softmax (row = i + 8*hl; reduce across 16-lane group) ----
#pragma unroll
        for (int i = 0; i < 8; ++i) {
            float mx = fmaxf(fmaxf(s[0][i], s[1][i]), fmaxf(s[2][i], s[3][i]));
            mx = fmaxf(mx, swz<SWZ_X1>(mx));
            mx = fmaxf(mx, swz<SWZ_X2>(mx));
            mx = fmaxf(mx, swz<SWZ_X4>(mx));
            mx = fmaxf(mx, swz<SWZ_X8>(mx));
            float mnew  = fmaxf(mrow[i], mx);
            float alpha = __expf(mrow[i] - mnew);
            float rs = 0.f;
#pragma unroll
            for (int ni = 0; ni < 4; ++ni) {
                float pv = __expf(s[ni][i] - mnew);
                s[ni][i] = pv;
                rs += pv;
            }
            rs = rs + swz<SWZ_X1>(rs);
            rs = rs + swz<SWZ_X2>(rs);
            rs = rs + swz<SWZ_X4>(rs);
            rs = rs + swz<SWZ_X8>(rs);
            lrow[i] = lrow[i] * alpha + rs;
            mrow[i] = mnew;
#pragma unroll
            for (int dn = 0; dn < 8; ++dn) o[dn][i] *= alpha;
        }

        // ---- P (C layout) -> LDS -> A-layout fragments ----
#pragma unroll
        for (int ni = 0; ni < 4; ++ni)
#pragma unroll
            for (int i = 0; i < 8; ++i)
                Ps[(wave * 16 + i + 8 * hl) * 72 + ni * 16 + laneN] = bf16_bits(s[ni][i]);
        __syncthreads();

        BF16Frag pf[2];
#pragma unroll
        for (int fr = 0; fr < 2; ++fr) {
            const unsigned short* p = &Ps[(wave * 16 + laneN) * 72 + fr * 32 + hl * 8];
            pf[fr].u[0] = *reinterpret_cast<const uint4*>(p);
            pf[fr].u[1] = *reinterpret_cast<const uint4*>(p + 16);
        }

        // ---- O += P * V : 8 d-tiles x 2 k-steps ----
#pragma unroll
        for (int dn = 0; dn < 8; ++dn) {
#pragma unroll
            for (int kf = 0; kf < 2; ++kf) {
                BF16Frag vb;
                const unsigned int* p = &Vs[(dn * 16 + laneN) * 36 + kf * 16 + hl * 4];
                vb.u[0] = *reinterpret_cast<const uint4*>(p);
                vb.u[1] = *reinterpret_cast<const uint4*>(p + 8);
                o[dn] = __builtin_amdgcn_wmma_f32_16x16x32_bf16(
                    false, pf[kf].v, false, vb.v, (short)0, o[dn], false, false);
            }
        }
        __syncthreads();
    }

    // ---- normalize + write [B,T,C] bf16 ----
#pragma unroll
    for (int dn = 0; dn < 8; ++dn)
#pragma unroll
        for (int i = 0; i < 8; ++i) {
            int tokr = q0 + wave * 16 + i + 8 * hl;
            float v = o[dn][i] / lrow[i];
            Out[((size_t)(b * 1024) + tokr) * 2048 + h * 128 + dn * 16 + laneN] = (__bf16)v;
        }
}

// ---------------------------------------------------------------------------
// Host-side orchestration
// ---------------------------------------------------------------------------
extern "C" void kernel_launch(void* const* d_in, const int* in_sizes, int n_in,
                              void* d_out, int out_size, void* d_ws, size_t ws_size,
                              hipStream_t stream) {
    (void)in_sizes; (void)n_in; (void)out_size; (void)ws_size;
    const float* x     = (const float*)d_in[0];
    const float* Wa    = (const float*)d_in[1];
    const float* Wp    = (const float*)d_in[2];
    const float* w_std = (const float*)d_in[3];
    const float* w_rec = (const float*)d_in[4];

    char* ws = (char*)d_ws;
    __bf16* xb    = (__bf16*)(ws);                 // x bf16            [4096][2048]
    __bf16* wbT   = (__bf16*)(ws + 16777216ull);   // Wb^T bf16         [6144][2048]
    __bf16* wpT   = (__bf16*)(ws + 41943040ull);   // Wp^T bf16         [2048][2048]
    __bf16* fused = (__bf16*)(ws + 50331648ull);   // fused bf16        [4096][6144]
    __bf16* Qfp   = (__bf16*)(ws + 100663296ull);  // Q routed          [B,H,T,128]
    __bf16* Kfp   = (__bf16*)(ws + 117440512ull);  // K routed          [B,H,T,128]
    __bf16* Vp    = (__bf16*)(ws + 134217728ull);  // V                 [B,H,T,128]
    __bf16* aout  = (__bf16*)(ws + 150994944ull);  // attn out          [4096][2048]

    // converts (Wb gets gate scales baked in + transpose to K-major)
    cvt_f32_to_bf16_v4<<<(4096 * 2048 / 4 + 255) / 256, 256, 0, stream>>>(x, xb, 4096 * 2048 / 4);
    cvt_transpose_bf16<true><<<dim3(6144 / 32, 2048 / 32), 256, 0, stream>>>(
        Wa, wbT, 2048, 6144, w_std, w_rec);
    cvt_transpose_bf16<false><<<dim3(2048 / 32, 2048 / 32), 256, 0, stream>>>(
        Wp, wpT, 2048, 2048, nullptr, nullptr);

    // fused = x @ Wb   [4096,2048]x[2048,6144] -> bf16
    gemm_wmma_bf16<false><<<dim3(6144 / 128, 4096 / 128), 256, 0, stream>>>(
        xb, wbT, (void*)fused, 4096, 6144, 2048);

    // split + per-head low-rank tail routing
    routing_kernel<<<4096, 256, 0, stream>>>(fused, w_rec, Qfp, Kfp, Vp);

    // causal flash attention
    flash_attn_kernel<<<dim3(16, 16, 4), 128, 0, stream>>>(Qfp, Kfp, Vp, aout);

    // y = attn_out @ Wp  [4096,2048]x[2048,2048] -> f32 d_out
    gemm_wmma_bf16<true><<<dim3(2048 / 128, 4096 / 128), 256, 0, stream>>>(
        aout, wpT, d_out, 4096, 2048, 2048);
}
